// EquiformerUnet_21354577396052
// MI455X (gfx1250) — compile-verified, matrix operand order checked by reading, not run on hardware
//
#include <hip/hip_runtime.h>
#include <math.h>

// ---------------- problem sizes ----------------
#define NN   8000
#define EE   96000
#define LL   16
#define CC   64
#define HH   4
#define NB_  256
#define EC_  48
#define FFN_ 128
#define CUTOFF (0.08f * 0.99f)

// ---------------- vector types ----------------
typedef __attribute__((ext_vector_type(16))) __bf16 bf16x16;
typedef __attribute__((ext_vector_type(8)))  __bf16 bf16x8;
typedef __attribute__((ext_vector_type(8)))  float  f32x8;

__device__ __forceinline__ f32x8 zero8() {
  f32x8 z = {0.f,0.f,0.f,0.f,0.f,0.f,0.f,0.f};
  return z;
}

__device__ __forceinline__ f32x8 wmma_bf16(bf16x16 a, bf16x16 b, f32x8 c) {
  return __builtin_amdgcn_wmma_f32_16x16x32_bf16(false, a, false, b, (short)0, c, false, false);
}

// Build an A fragment (16x32 bf16 tile, K-chunk kc) from a row-major bf16
// matrix with row stride `stride` (in elements). ISA 7.12.2 layout:
// lane&15 = row M, lanes<16 hold K = kc*32 + {0..7, 16..23},
// lanes>=16 hold K = kc*32 + {8..15, 24..31}.
__device__ __forceinline__ bf16x16 load_a_frag(const __bf16* base, int stride,
                                               int lane, int kc) {
  int row = lane & 15;
  int kb  = (lane >> 4) * 8;
  const __bf16* p = base + row * stride + kc * 32 + kb;
  bf16x8 lo = *(const bf16x8*)(p);
  bf16x8 hi = *(const bf16x8*)(p + 16);
  bf16x16 a;
#pragma unroll
  for (int i = 0; i < 8; ++i) { a[i] = lo[i]; a[i + 8] = hi[i]; }
  return a;
}

// A fragment built from f32 data (converted on the fly).
__device__ __forceinline__ bf16x16 load_a_frag_f32(const float* base, int stride,
                                                   int lane, int kc) {
  int row = lane & 15;
  int kb  = (lane >> 4) * 8;
  const float* p = base + row * stride + kc * 32 + kb;
  bf16x16 a;
#pragma unroll
  for (int i = 0; i < 8; ++i) { a[i] = (__bf16)p[i]; a[i + 8] = (__bf16)p[i + 16]; }
  return a;
}

// Packed B fragment: weights pre-permuted so each (kc, ntile, lane) owns 16
// contiguous bf16 = one 32B load.
__device__ __forceinline__ bf16x16 load_b_frag(const __bf16* packed, int ntiles,
                                               int kc, int t, int lane) {
  return *(const bf16x16*)(packed + (((size_t)kc * ntiles + t) * 32 + lane) * 16);
}

// ---------------- weight packing f32 -> bf16 B-fragment layout ----------------
// Layout: [kc][ntile][lane][16]; lane&15 = column within tile, lane>>4 picks
// the K half (kb = 8*(lane>>4)); elems 0..7 = K kb+j, 8..15 = K kb+16+j.
__global__ void pack_weights_kernel(const float* __restrict__ W,
                                    __bf16* __restrict__ Wp, int K, int Ncols) {
  int kcs = K >> 5, nts = Ncols >> 4;
  int idx = blockIdx.x * blockDim.x + threadIdx.x;
  if (idx >= kcs * nts * 32) return;
  int lane = idx & 31;
  int t    = (idx >> 5) % nts;
  int kc   = idx / (32 * nts);
  int col  = t * 16 + (lane & 15);
  int kb   = kc * 32 + ((lane >> 4) * 8);
  __bf16* out = Wp + (size_t)idx * 16;
#pragma unroll
  for (int j = 0; j < 8; ++j) out[j]     = (__bf16)W[(kb + j) * Ncols + col];
#pragma unroll
  for (int j = 0; j < 8; ++j) out[8 + j] = (__bf16)W[(kb + 16 + j) * Ncols + col];
}

// ---------------- RMS-norm over (L,C)=1024 per node, f32 -> bf16 ----------------
__global__ void rmsnorm_bf16_kernel(const float* __restrict__ x,
                                    __bf16* __restrict__ out) {
  __shared__ float sh[256];
  int node = blockIdx.x;
  const float* xp = x + (size_t)node * (LL * CC);
  float s = 0.f;
  for (int i = threadIdx.x; i < LL * CC; i += 256) { float v = xp[i]; s += v * v; }
  sh[threadIdx.x] = s;
  __syncthreads();
  for (int off = 128; off > 0; off >>= 1) {
    if (threadIdx.x < off) sh[threadIdx.x] += sh[threadIdx.x + off];
    __syncthreads();
  }
  float r = rsqrtf(sh[0] * (1.0f / (LL * CC)) + 1e-6f);
  __bf16* op = out + (size_t)node * (LL * CC);
  for (int i = threadIdx.x; i < LL * CC; i += 256) op[i] = (__bf16)(xp[i] * r);
}

// ---------------- init segment-max (bits of -inf) and denom ----------------
__global__ void init_stats_kernel(unsigned int* __restrict__ mmax,
                                  float* __restrict__ denom, int n) {
  int i = blockIdx.x * blockDim.x + threadIdx.x;
  if (i < n) { mmax[i] = 0xFF800000u; denom[i] = 0.f; }
}

__device__ __forceinline__ void atomic_max_f32(unsigned int* addr, float val) {
  if (val >= 0.f) atomicMax((int*)addr, __float_as_int(val));
  else            atomicMin(addr, __float_as_uint(val));
}

// ---------------- per-edge invariant path (RBF MLP, e_c, s0, logits) ----------
// One wave per edge, 8 edges per 256-thread block. VALU + LDS; WMMA is spent
// on the big GEMMs elsewhere.
__global__ void edge_invariant_kernel(const float* __restrict__ pos,
                                      const __bf16* __restrict__ xn,
                                      const int* __restrict__ edge_index,
                                      const float* __restrict__ W1,
                                      const float* __restrict__ b1,
                                      const float* __restrict__ W2,
                                      const float* __restrict__ b2,
                                      const float* __restrict__ W_edge,
                                      const float* __restrict__ W_src,
                                      const float* __restrict__ W_dst,
                                      const float* __restrict__ W_alpha,
                                      const float* __restrict__ v_alpha,
                                      float* __restrict__ e_c,
                                      float* __restrict__ logits,
                                      unsigned int* __restrict__ mmax) {
  __shared__ float sh_e[8][NB_];
  __shared__ float sh_h1[8][EC_];
  __shared__ float sh_h2[8][EC_];
  __shared__ float sh_ec[8][CC];
  __shared__ float sh_x[8][2 * CC];
  __shared__ float sh_s0[8][CC];
  __shared__ float sh_a[8][HH * 64];
  int wave = threadIdx.x >> 5, lane = threadIdx.x & 31;
  int e = blockIdx.x * 8 + wave;
  int src = edge_index[e], dst = edge_index[EE + e];

  float dx = pos[src * 3 + 0] - pos[dst * 3 + 0];
  float dy = pos[src * 3 + 1] - pos[dst * 3 + 1];
  float dz = pos[src * 3 + 2] - pos[dst * 3 + 2];
  float d  = sqrtf(dx * dx + dy * dy + dz * dz + 1e-12f);
  float u  = fminf(fmaxf(d / CUTOFF, 0.f), 1.f);
  float env = 0.5f * (cosf(3.14159265358979f * u) + 1.f);
  float inv_std = (float)NB_ / CUTOFF;
  for (int j = lane; j < NB_; j += 32) {
    float c = CUTOFF * ((float)j / (float)(NB_ - 1));
    float t = (d - c) * inv_std;
    sh_e[wave][j] = __expf(-0.5f * t * t) * env;
  }
  // MLP1: NB -> EC, silu
  for (int j = lane; j < EC_; j += 32) {
    float acc = b1[j];
    for (int k = 0; k < NB_; ++k) acc += sh_e[wave][k] * W1[k * EC_ + j];
    sh_h1[wave][j] = acc / (1.f + __expf(-acc));
  }
  // MLP2: EC -> EC, silu
  for (int j = lane; j < EC_; j += 32) {
    float acc = b2[j];
    for (int k = 0; k < EC_; ++k) acc += sh_h1[wave][k] * W2[k * EC_ + j];
    sh_h2[wave][j] = acc / (1.f + __expf(-acc));
  }
  // e_c: EC -> C
  for (int c = lane; c < CC; c += 32) {
    float acc = 0.f;
    for (int k = 0; k < EC_; ++k) acc += sh_h2[wave][k] * W_edge[k * CC + c];
    sh_ec[wave][c] = acc;
    e_c[(size_t)e * CC + c] = acc;
  }
  // l=0 rows of normed features
  for (int c = lane; c < CC; c += 32) {
    sh_x[wave][c]      = (float)xn[(size_t)src * (LL * CC) + c];
    sh_x[wave][CC + c] = (float)xn[(size_t)dst * (LL * CC) + c];
  }
  // s0 = (xn0_src @ W_src + xn0_dst @ W_dst) * e_c
  for (int c = lane; c < CC; c += 32) {
    float acc = 0.f;
    for (int k = 0; k < CC; ++k)
      acc += sh_x[wave][k] * W_src[k * CC + c] + sh_x[wave][CC + k] * W_dst[k * CC + c];
    sh_s0[wave][c] = acc * sh_ec[wave][c];
  }
  // a = leaky_relu(s0 @ W_alpha, 0.2)
  for (int j = lane; j < HH * 64; j += 32) {
    float acc = 0.f;
    for (int k = 0; k < CC; ++k) acc += sh_s0[wave][k] * W_alpha[k * (HH * 64) + j];
    sh_a[wave][j] = acc > 0.f ? acc : 0.2f * acc;
  }
  // logits[h] = dot(a[h,:], v_alpha[h,:]); seed segment-max
  if (lane < HH) {
    int h = lane;
    float acc = 0.f;
    for (int k = 0; k < 64; ++k) acc += sh_a[wave][h * 64 + k] * v_alpha[h * 64 + k];
    logits[(size_t)e * HH + h] = acc;
    atomic_max_f32(&mmax[(size_t)dst * HH + h], acc);
  }
}

// ---------------- edge softmax: w = exp(l - m), accumulate denom -------------
__global__ void edge_softmax_kernel(float* __restrict__ logits,
                                    const unsigned int* __restrict__ mmax,
                                    const int* __restrict__ edge_index,
                                    float* __restrict__ denom) {
  int i = blockIdx.x * blockDim.x + threadIdx.x;
  if (i >= EE * HH) return;
  int e = i >> 2, h = i & 3;
  int dst = edge_index[EE + e];
  float m = __uint_as_float(mmax[(size_t)dst * HH + h]);
  float w = __expf(logits[i] - m);
  logits[i] = w;
  atomicAdd(&denom[(size_t)dst * HH + h], w);
}

// ---------------- WMMA edge message + value + scatter ------------------------
// One wave per edge: msg(16x64) = xn[src]@W_src + xn[dst]@W_dst (16 WMMAs),
// modulate l=0 by e_c, restage via LDS, v = msg@W_v (8 WMMAs), scale by alpha,
// atomic scatter into agg[dst].
__global__ void __launch_bounds__(256)
edge_message_kernel(const __bf16* __restrict__ xn,
                    const int* __restrict__ edge_index,
                    const float* __restrict__ e_c,
                    const float* __restrict__ wbuf,
                    const float* __restrict__ denom,
                    const __bf16* __restrict__ Wsrc_p,
                    const __bf16* __restrict__ Wdst_p,
                    const __bf16* __restrict__ Wv_p,
                    float* __restrict__ agg) {
  __shared__ __align__(16) __bf16 stage[8][LL * CC];
  int wave = threadIdx.x >> 5, lane = threadIdx.x & 31;
  int e = blockIdx.x * 8 + wave;
  int src = edge_index[e], dst = edge_index[EE + e];
  const __bf16* As = xn + (size_t)src * (LL * CC);
  const __bf16* Ad = xn + (size_t)dst * (LL * CC);

  f32x8 acc[4];
#pragma unroll
  for (int t = 0; t < 4; ++t) acc[t] = zero8();
#pragma unroll
  for (int kc = 0; kc < 2; ++kc) {
    bf16x16 as = load_a_frag(As, CC, lane, kc);
    bf16x16 ad = load_a_frag(Ad, CC, lane, kc);
#pragma unroll
    for (int t = 0; t < 4; ++t) {
      acc[t] = wmma_bf16(as, load_b_frag(Wsrc_p, 4, kc, t, lane), acc[t]);
      acc[t] = wmma_bf16(ad, load_b_frag(Wdst_p, 4, kc, t, lane), acc[t]);
    }
  }
  int n  = lane & 15;
  int mb = (lane >> 4) * 8;           // D layout: vgpr r -> row mb + r
  // modulate l=0 (row m==0 lives in vgpr0 of lanes 0..15)
#pragma unroll
  for (int t = 0; t < 4; ++t) {
    float f = (lane < 16) ? e_c[(size_t)e * CC + t * 16 + n] : 1.0f;
    acc[t][0] *= f;
  }
  // restage message (bf16, row-major 16x64) for use as next A operand
  __bf16* st = stage[wave];
#pragma unroll
  for (int t = 0; t < 4; ++t)
#pragma unroll
    for (int r = 0; r < 8; ++r)
      st[(mb + r) * CC + t * 16 + n] = (__bf16)acc[t][r];

  f32x8 vacc[4];
#pragma unroll
  for (int t = 0; t < 4; ++t) vacc[t] = zero8();
#pragma unroll
  for (int kc = 0; kc < 2; ++kc) {
    bf16x16 am = load_a_frag(st, CC, lane, kc);
#pragma unroll
    for (int t = 0; t < 4; ++t)
      vacc[t] = wmma_bf16(am, load_b_frag(Wv_p, 4, kc, t, lane), vacc[t]);
  }
  // alpha scale (head h == n-tile t since H*VC = 4*16) + scatter
#pragma unroll
  for (int t = 0; t < 4; ++t) {
    float alpha = wbuf[(size_t)e * HH + t] / (denom[(size_t)dst * HH + t] + 1e-9f);
#pragma unroll
    for (int r = 0; r < 8; ++r)
      atomicAdd(&agg[((size_t)dst * LL + mb + r) * CC + t * 16 + n],
                vacc[t][r] * alpha);
  }
}

// ---------------- x1 = x + agg @ W_o (WMMA) ----------------------------------
__global__ void __launch_bounds__(256)
wo_residual_kernel(const float* __restrict__ agg,
                   const float* __restrict__ x,
                   const __bf16* __restrict__ Wo_p,
                   float* __restrict__ x1) {
  int wave = threadIdx.x >> 5, lane = threadIdx.x & 31;
  int node = blockIdx.x * 8 + wave;
  const float* ap = agg + (size_t)node * (LL * CC);
  f32x8 acc[4];
#pragma unroll
  for (int t = 0; t < 4; ++t) acc[t] = zero8();
#pragma unroll
  for (int kc = 0; kc < 2; ++kc) {
    bf16x16 a = load_a_frag_f32(ap, CC, lane, kc);
#pragma unroll
    for (int t = 0; t < 4; ++t)
      acc[t] = wmma_bf16(a, load_b_frag(Wo_p, 4, kc, t, lane), acc[t]);
  }
  int n = lane & 15, mb = (lane >> 4) * 8;
#pragma unroll
  for (int t = 0; t < 4; ++t)
#pragma unroll
    for (int r = 0; r < 8; ++r) {
      size_t idx = ((size_t)node * LL + mb + r) * CC + t * 16 + n;
      x1[idx] = x[idx] + acc[t][r];
    }
}

// ---------------- FFN: gate GEMV + hidden/out WMMA GEMMs, fused residual -----
__global__ void __launch_bounds__(256)
ffn_kernel(const __bf16* __restrict__ y,
           const float* __restrict__ x1,
           const float* __restrict__ W_gate,
           const __bf16* __restrict__ Wh_p,
           const __bf16* __restrict__ Wf_p,
           float* __restrict__ out) {
  __shared__ float g_sh[8][FFN_];
  __shared__ __align__(16) __bf16 h_sh[8][LL * FFN_];
  int wave = threadIdx.x >> 5, lane = threadIdx.x & 31;
  int node = blockIdx.x * 8 + wave;
  const __bf16* yp = y + (size_t)node * (LL * CC);

  // gate: g = silu(y0 @ W_gate)
  for (int j = lane; j < FFN_; j += 32) {
    float acc = 0.f;
    for (int c = 0; c < CC; ++c) acc += (float)yp[c] * W_gate[c * FFN_ + j];
    g_sh[wave][j] = acc / (1.f + __expf(-acc));
  }

  // h = (y @ W_hidden) * g : 16x64 @ 64x128
  f32x8 h[8];
#pragma unroll
  for (int t = 0; t < 8; ++t) h[t] = zero8();
#pragma unroll
  for (int kc = 0; kc < 2; ++kc) {
    bf16x16 a = load_a_frag(yp, CC, lane, kc);
#pragma unroll
    for (int t = 0; t < 8; ++t)
      h[t] = wmma_bf16(a, load_b_frag(Wh_p, 8, kc, t, lane), h[t]);
  }
  int n = lane & 15, mb = (lane >> 4) * 8;
  __bf16* st = h_sh[wave];
#pragma unroll
  for (int t = 0; t < 8; ++t) {
    float gv = g_sh[wave][t * 16 + n];
#pragma unroll
    for (int r = 0; r < 8; ++r)
      st[(mb + r) * FFN_ + t * 16 + n] = (__bf16)(h[t][r] * gv);
  }

  // out = x1 + h @ W_ffn_out : 16x128 @ 128x64
  f32x8 o[4];
#pragma unroll
  for (int t = 0; t < 4; ++t) o[t] = zero8();
#pragma unroll
  for (int kc = 0; kc < 4; ++kc) {
    bf16x16 a = load_a_frag(st, FFN_, lane, kc);
#pragma unroll
    for (int t = 0; t < 4; ++t)
      o[t] = wmma_bf16(a, load_b_frag(Wf_p, 4, kc, t, lane), o[t]);
  }
#pragma unroll
  for (int t = 0; t < 4; ++t)
#pragma unroll
    for (int r = 0; r < 8; ++r) {
      size_t idx = ((size_t)node * LL + mb + r) * CC + t * 16 + n;
      out[idx] = x1[idx] + o[t][r];
    }
}

// ---------------- host launch ----------------
extern "C" void kernel_launch(void* const* d_in, const int* in_sizes, int n_in,
                              void* d_out, int out_size, void* d_ws, size_t ws_size,
                              hipStream_t stream) {
  const float* pos      = (const float*)d_in[0];
  const float* x        = (const float*)d_in[1];
  const float* W_src    = (const float*)d_in[2];
  const float* W_dst    = (const float*)d_in[3];
  const float* W1_rbf   = (const float*)d_in[4];
  const float* b1_rbf   = (const float*)d_in[5];
  const float* W2_rbf   = (const float*)d_in[6];
  const float* b2_rbf   = (const float*)d_in[7];
  const float* W_edge   = (const float*)d_in[8];
  const float* W_alpha  = (const float*)d_in[9];
  const float* v_alpha  = (const float*)d_in[10];
  const float* W_v      = (const float*)d_in[11];
  const float* W_o      = (const float*)d_in[12];
  const float* W_gate   = (const float*)d_in[13];
  const float* W_hidden = (const float*)d_in[14];
  const float* W_ffnout = (const float*)d_in[15];
  const int*   edge_idx = (const int*)d_in[16];
  float* outp = (float*)d_out;

  char* ws = (char*)d_ws;
  size_t off = 0;
  auto alloc = [&](size_t bytes) -> void* {
    off = (off + 255) & ~(size_t)255;
    void* p = ws + off;
    off += bytes;
    return p;
  };
  __bf16* xn_bf   = (__bf16*)alloc((size_t)NN * LL * CC * 2);
  __bf16* y_bf    = (__bf16*)alloc((size_t)NN * LL * CC * 2);
  float*  e_c     = (float*) alloc((size_t)EE * CC * 4);
  float*  wbuf    = (float*) alloc((size_t)EE * HH * 4);   // logits, then softmax w
  unsigned int* mmax = (unsigned int*)alloc((size_t)NN * HH * 4);
  float*  denom   = (float*) alloc((size_t)NN * HH * 4);
  float*  agg     = (float*) alloc((size_t)NN * LL * CC * 4);
  __bf16* Wsrc_p  = (__bf16*)alloc(2 * 4 * 32 * 16 * 2);
  __bf16* Wdst_p  = (__bf16*)alloc(2 * 4 * 32 * 16 * 2);
  __bf16* Wv_p    = (__bf16*)alloc(2 * 4 * 32 * 16 * 2);
  __bf16* Wo_p    = (__bf16*)alloc(2 * 4 * 32 * 16 * 2);
  __bf16* Wh_p    = (__bf16*)alloc(2 * 8 * 32 * 16 * 2);
  __bf16* Wf_p    = (__bf16*)alloc(4 * 4 * 32 * 16 * 2);

  // pack weights (bf16, B-fragment layout)
  pack_weights_kernel<<<1, 256, 0, stream>>>(W_src,    Wsrc_p, 64, 64);
  pack_weights_kernel<<<1, 256, 0, stream>>>(W_dst,    Wdst_p, 64, 64);
  pack_weights_kernel<<<1, 256, 0, stream>>>(W_v,      Wv_p,   64, 64);
  pack_weights_kernel<<<1, 256, 0, stream>>>(W_o,      Wo_p,   64, 64);
  pack_weights_kernel<<<2, 256, 0, stream>>>(W_hidden, Wh_p,   64, 128);
  pack_weights_kernel<<<4, 256, 0, stream>>>(W_ffnout, Wf_p,   128, 64);

  // init stats + zero agg
  init_stats_kernel<<<(NN * HH + 255) / 256, 256, 0, stream>>>(mmax, denom, NN * HH);
  hipMemsetAsync(agg, 0, (size_t)NN * LL * CC * 4, stream);

  // xn = rmsnorm(x) -> bf16
  rmsnorm_bf16_kernel<<<NN, 256, 0, stream>>>(x, xn_bf);

  // per-edge invariants + logits + segment max
  edge_invariant_kernel<<<EE / 8, 256, 0, stream>>>(
      pos, xn_bf, edge_idx, W1_rbf, b1_rbf, W2_rbf, b2_rbf, W_edge,
      W_src, W_dst, W_alpha, v_alpha, e_c, wbuf, mmax);

  // softmax weights + denom
  edge_softmax_kernel<<<(EE * HH + 255) / 256, 256, 0, stream>>>(
      wbuf, mmax, edge_idx, denom);

  // WMMA edge message + value + scatter
  edge_message_kernel<<<EE / 8, 256, 0, stream>>>(
      xn_bf, edge_idx, e_c, wbuf, denom, Wsrc_p, Wdst_p, Wv_p, agg);

  // x1 = x + agg @ W_o  (written into d_out)
  wo_residual_kernel<<<NN / 8, 256, 0, stream>>>(agg, x, Wo_p, outp);

  // y = rmsnorm(x1) -> bf16
  rmsnorm_bf16_kernel<<<NN, 256, 0, stream>>>(outp, y_bf);

  // out = x1 + ffn(y)
  ffn_kernel<<<NN / 8, 256, 0, stream>>>(y_bf, outp, W_gate, Wh_p, Wf_p, outp);

  (void)in_sizes; (void)n_in; (void)out_size; (void)ws_size;
}
